// ScaledDotAttention_6502580486602
// MI455X (gfx1250) — matrix operand order repaired and checked by reading, not compiled, not run on hardware
//
#include <hip/hip_runtime.h>
#include <stdint.h>
#include <stddef.h>

#define HIDDEN 1024
#define BATCH  4
#define SEQ    4096
#define NKQ    4096

typedef __attribute__((ext_vector_type(16))) __bf16 v16bf;
typedef __attribute__((ext_vector_type(8)))  __bf16 v8bf;
typedef __attribute__((ext_vector_type(8)))  float  v8f;
typedef int v4i32 __attribute__((vector_size(16)));  // matches builtin proto

#if __has_builtin(__builtin_amdgcn_global_load_async_to_lds_b128)
#define HAVE_ASYNC 1
#else
#define HAVE_ASYNC 0
#endif

template <int N>
__device__ __forceinline__ void wait_async_le() {
#if HAVE_ASYNC
#if __has_builtin(__builtin_amdgcn_s_wait_asynccnt)
  __builtin_amdgcn_s_wait_asynccnt(N);
#else
  asm volatile("s_wait_asynccnt %0" ::"i"(N) : "memory");
#endif
#endif
}

// 16-byte global -> LDS copy; async CDNA5 DMA path when available.
__device__ __forceinline__ void copy16(const void* g, void* l) {
#if HAVE_ASYNC
  __builtin_amdgcn_global_load_async_to_lds_b128(
      (__attribute__((address_space(1))) v4i32*)g,
      (__attribute__((address_space(3))) v4i32*)l, 0, 0);
#else
  *(uint4*)l = *(const uint4*)g;
#endif
}

__device__ __forceinline__ v16bf make_v16(v8bf lo, v8bf hi) {
  return __builtin_shufflevector(lo, hi, 0, 1, 2, 3, 4, 5, 6, 7, 8, 9, 10, 11,
                                 12, 13, 14, 15);
}

// A fragment: 16x32 bf16, LDS tile row-major [row][32], lda == 32.
// lanes 0-15: row=lane, K = {0..7, 16..23}; lanes 16-31: row=lane-16, K = {8..15, 24..31}
__device__ __forceinline__ v16bf load_a_frag(const __bf16* As, int row0, int lane) {
  const int r = row0 + (lane & 15);
  const int hi = (lane >> 4) & 1;
  const __bf16* p = As + r * 32;
  v8bf x0 = *(const v8bf*)(p + hi * 8);
  v8bf x1 = *(const v8bf*)(p + 16 + hi * 8);
  return make_v16(x0, x1);
}

// B fragment: 32x16 bf16 from a B^T-row-major LDS tile [n][32], ldb == 32.
// lane n = lane&15; lanes 0-15: K = {0..7, 16..23}; lanes 16-31: K = {8..15, 24..31}
__device__ __forceinline__ v16bf load_b_frag(const __bf16* Bs, int col0, int lane) {
  const int n = col0 + (lane & 15);
  const int hi = (lane >> 4) & 1;
  const __bf16* p = Bs + n * 32;
  v8bf x0 = *(const v8bf*)(p + hi * 8);
  v8bf x1 = *(const v8bf*)(p + 16 + hi * 8);
  return make_v16(x0, x1);
}

__device__ __forceinline__ v8f wmma_bf16(v16bf a, v16bf b, v8f c) {
  return __builtin_amdgcn_wmma_f32_16x16x32_bf16(false, a, false, b, (short)0,
                                                 c, false, false);
}

// One BK=32 step: wave computes its 32x64 sub-tile (2x4 accumulators,
// 6 fragment loads -> 8 WMMAs).
__device__ __forceinline__ void mma_step(const __bf16* ldsA, const __bf16* ldsB,
                                         int mwBase, int nwBase, int lane,
                                         v8f acc[2][4]) {
  v16bf a0 = load_a_frag(ldsA, mwBase, lane);
  v16bf a1 = load_a_frag(ldsA, mwBase + 16, lane);
  v16bf b[4];
#pragma unroll
  for (int j = 0; j < 4; ++j) b[j] = load_b_frag(ldsB, nwBase + 16 * j, lane);
#pragma unroll
  for (int j = 0; j < 4; ++j) {
    acc[0][j] = wmma_bf16(a0, b[j], acc[0][j]);
    acc[1][j] = wmma_bf16(a1, b[j], acc[1][j]);
  }
}

// ---------------- Kernel 1: Y_bf16 = X_f32 @ W_f32^T + bias ----------------
// X: [M,K] row-major, W: [N,K] row-major (torch Linear), Y: [M,N] bf16.
// Block tile 128x128, BK=32; 8 waves in 4x2; per-wave 32x64.
__global__ __launch_bounds__(256) void proj_gemm_bf16_kernel(
    const float* __restrict__ X, const float* __restrict__ W,
    const float* __restrict__ bias, __bf16* __restrict__ Y, int M, int N,
    int K) {
  __shared__ __align__(16) __bf16 ldsA[128 * 32];
  __shared__ __align__(16) __bf16 ldsB[128 * 32];
  const int tid = threadIdx.x;
  const int lane = tid & 31;
  const int wave = tid >> 5;
  const int mwBase = (wave >> 1) * 32;  // 4 wave rows
  const int nwBase = (wave & 1) * 64;   // 2 wave cols
  const int m0 = blockIdx.y * 128;
  const int n0 = blockIdx.x * 128;

  v8f acc[2][4] = {};
  for (int kc = 0; kc < K; kc += 32) {
    __syncthreads();
#pragma unroll
    for (int i = 0; i < 16; ++i) {  // A: 128x32, convert f32 -> bf16
      int idx = tid + i * 256;
      int m = idx >> 5, k = idx & 31;
      ldsA[idx] = (__bf16)X[(size_t)(m0 + m) * K + kc + k];
    }
#pragma unroll
    for (int i = 0; i < 16; ++i) {  // B^T tile: ldsB[n][k] = W[n0+n][kc+k]
      int idx = tid + i * 256;
      int n = idx >> 5, k = idx & 31;
      ldsB[idx] = (__bf16)W[(size_t)(n0 + n) * K + kc + k];
    }
    __syncthreads();
    mma_step(ldsA, ldsB, mwBase, nwBase, lane, acc);
  }

  const int nn = lane & 15;
  const int mr = (lane >> 4) * 8;
#pragma unroll
  for (int i = 0; i < 2; ++i)
#pragma unroll
    for (int j = 0; j < 4; ++j) {
      const int col = n0 + nwBase + j * 16 + nn;
      const float bv = bias[col];
      const int row = m0 + mwBase + i * 16 + mr;
#pragma unroll
      for (int r = 0; r < 8; ++r)
        Y[(size_t)(row + r) * N + col] = (__bf16)(acc[i][j][r] + bv);
    }
}

// ------------- Kernel 2: scores = (K_bf16 @ Q_bf16^T) * scale --------------
// Both operands bf16 with K-dim contiguous -> async DMA to double-buffered
// LDS, overlapped with WMMA via s_wait_asynccnt.
__device__ __forceinline__ void scores_issue_copies(
    const __bf16* __restrict__ Kbase, const __bf16* __restrict__ Qbase,
    __bf16* lA, __bf16* lB, int m0, int n0, int kc, int tid) {
#pragma unroll
  for (int i = 0; i < 2; ++i) {  // 128 rows x 64B each for A and B
    const int idx = tid + i * 256;
    const int row = idx >> 2;
    const int seg = (idx & 3) * 16;
    copy16((const char*)(Kbase + (size_t)(m0 + row) * HIDDEN + kc) + seg,
           (char*)lA + row * 64 + seg);
    copy16((const char*)(Qbase + (size_t)(n0 + row) * HIDDEN + kc) + seg,
           (char*)lB + row * 64 + seg);
  }
}

__global__ __launch_bounds__(256) void scores_gemm_kernel(
    const __bf16* __restrict__ Kb, const __bf16* __restrict__ Qb,
    float* __restrict__ attn, float scale) {
  __shared__ __align__(16) __bf16 ldsA[2][128 * 32];
  __shared__ __align__(16) __bf16 ldsB[2][128 * 32];
  const int tid = threadIdx.x;
  const int lane = tid & 31;
  const int wave = tid >> 5;
  const int mwBase = (wave >> 1) * 32;
  const int nwBase = (wave & 1) * 64;
  const int b = blockIdx.z;
  const int m0 = blockIdx.y * 128;  // S tile
  const int n0 = blockIdx.x * 128;  // Kq tile
  const __bf16* Kbase = Kb + (size_t)b * SEQ * HIDDEN;
  const __bf16* Qbase = Qb + (size_t)b * NKQ * HIDDEN;

  v8f acc[2][4] = {};
  constexpr int NCHUNK = HIDDEN / 32;
  scores_issue_copies(Kbase, Qbase, ldsA[0], ldsB[0], m0, n0, 0, tid);
  for (int ic = 0; ic < NCHUNK; ++ic) {
    const int cur = ic & 1;
    __syncthreads();  // everyone done reading buf cur^1 (iteration ic-1)
    if (ic + 1 < NCHUNK) {
      scores_issue_copies(Kbase, Qbase, ldsA[cur ^ 1], ldsB[cur ^ 1], m0, n0,
                          (ic + 1) * 32, tid);
      wait_async_le<4>();  // 4 fresh copies in flight; buf `cur` is complete
    } else {
      wait_async_le<0>();
    }
    __syncthreads();  // buf `cur` visible to all waves
    mma_step(ldsA[cur], ldsB[cur], mwBase, nwBase, lane, acc);
  }

  float* Obase = attn + (size_t)b * SEQ * NKQ;
  const int nn = lane & 15;
  const int mr = (lane >> 4) * 8;
#pragma unroll
  for (int i = 0; i < 2; ++i)
#pragma unroll
    for (int j = 0; j < 4; ++j) {
      const int col = n0 + nwBase + j * 16 + nn;
      const int row = m0 + mwBase + i * 16 + mr;
#pragma unroll
      for (int r = 0; r < 8; ++r)
        Obase[(size_t)(row + r) * NKQ + col] = acc[i][j][r] * scale;
    }
}

// -------- Kernel 3: softmax over dim S (axis=1) per (b, kq) column ---------
__global__ __launch_bounds__(256) void softmax_col_kernel(
    float* __restrict__ attn) {
  const int b = blockIdx.y;
  const int col = blockIdx.x * 256 + threadIdx.x;
  float* base = attn + (size_t)b * SEQ * NKQ + col;
  float m = -3.402823466e+38f;
  for (int s = 0; s < SEQ; ++s) m = fmaxf(m, base[(size_t)s * NKQ]);
  float sum = 0.f;
  for (int s = 0; s < SEQ; ++s) {
    const float e = __expf(base[(size_t)s * NKQ] - m);
    base[(size_t)s * NKQ] = e;
    sum += e;
  }
  const float inv = 1.0f / sum;
  for (int s = 0; s < SEQ; ++s) base[(size_t)s * NKQ] *= inv;
}

// -------- Kernel 4: context[b,kq,h] = sum_s attn[b,s,kq] * v[b,s,h] --------
// M=KQ, N=HIDDEN, K=SEQ. A is attn^T (fp32 -> bf16), B^T tile from V rows.
__global__ __launch_bounds__(256) void context_gemm_kernel(
    const float* __restrict__ attn, const __bf16* __restrict__ Vb,
    float* __restrict__ ctx) {
  __shared__ __align__(16) __bf16 ldsA[128 * 32];
  __shared__ __align__(16) __bf16 ldsB[128 * 32];
  const int tid = threadIdx.x;
  const int lane = tid & 31;
  const int wave = tid >> 5;
  const int mwBase = (wave >> 1) * 32;
  const int nwBase = (wave & 1) * 64;
  const int b = blockIdx.z;
  const int m0 = blockIdx.y * 128;  // KQ tile
  const int n0 = blockIdx.x * 128;  // H tile
  const float* Abase = attn + (size_t)b * SEQ * NKQ;
  const __bf16* Vbase = Vb + (size_t)b * SEQ * HIDDEN;

  v8f acc[2][4] = {};
  for (int kc = 0; kc < SEQ; kc += 32) {
    __syncthreads();
#pragma unroll
    for (int i = 0; i < 16; ++i) {  // ldsA[m][k] = attn[kc+k][m0+m]
      int idx = tid + i * 256;
      int m = idx & 127, k = idx >> 7;
      ldsA[m * 32 + k] = (__bf16)Abase[(size_t)(kc + k) * NKQ + m0 + m];
    }
#pragma unroll
    for (int i = 0; i < 16; ++i) {  // ldsB[n][k] = V[kc+k][n0+n]
      int idx = tid + i * 256;
      int n = idx & 127, k = idx >> 7;
      ldsB[n * 32 + k] = Vbase[(size_t)(kc + k) * HIDDEN + n0 + n];
    }
    __syncthreads();
    mma_step(ldsA, ldsB, mwBase, nwBase, lane, acc);
  }

  float* Obase = ctx + (size_t)b * NKQ * HIDDEN;
  const int nn = lane & 15;
  const int mr = (lane >> 4) * 8;
#pragma unroll
  for (int i = 0; i < 2; ++i)
#pragma unroll
    for (int j = 0; j < 4; ++j) {
      const int col = n0 + nwBase + j * 16 + nn;
      const int row = m0 + mwBase + i * 16 + mr;
#pragma unroll
      for (int r = 0; r < 8; ++r)
        Obase[(size_t)(row + r) * HIDDEN + col] = acc[i][j][r];
    }
}

extern "C" void kernel_launch(void* const* d_in, const int* in_sizes, int n_in,
                              void* d_out, int out_size, void* d_ws,
                              size_t ws_size, hipStream_t stream) {
  (void)in_sizes; (void)n_in; (void)out_size; (void)ws_size;
  const float* queries = (const float*)d_in[0];
  const float* keys    = (const float*)d_in[1];
  const float* values  = (const float*)d_in[2];
  const float* Wq = (const float*)d_in[3];
  const float* bq = (const float*)d_in[4];
  const float* Wk = (const float*)d_in[5];
  const float* bk = (const float*)d_in[6];
  const float* Wv = (const float*)d_in[7];
  const float* bv = (const float*)d_in[8];

  float* out = (float*)d_out;
  float* ctx = out;                                  // [B,KQ,H]
  float* attn = out + (size_t)BATCH * NKQ * HIDDEN;  // [B,S,KQ]

  __bf16* qb = (__bf16*)d_ws;                        // [B*KQ, H] bf16
  __bf16* kb = qb + (size_t)BATCH * NKQ * HIDDEN;    // [B*S, H]  bf16
  __bf16* vb = kb + (size_t)BATCH * SEQ * HIDDEN;    // [B*S, H]  bf16

  const int Mflat = BATCH * NKQ;  // batch rows concatenate for projections
  dim3 blk(256);
  dim3 gproj(HIDDEN / 128, Mflat / 128);
  proj_gemm_bf16_kernel<<<gproj, blk, 0, stream>>>(queries, Wq, bq, qb, Mflat,
                                                   HIDDEN, HIDDEN);
  proj_gemm_bf16_kernel<<<gproj, blk, 0, stream>>>(keys, Wk, bk, kb, Mflat,
                                                   HIDDEN, HIDDEN);
  proj_gemm_bf16_kernel<<<gproj, blk, 0, stream>>>(values, Wv, bv, vb, Mflat,
                                                   HIDDEN, HIDDEN);

  dim3 gsc(NKQ / 128, SEQ / 128, BATCH);
  scores_gemm_kernel<<<gsc, blk, 0, stream>>>(kb, qb, attn, 0.03125f);

  dim3 gsm(NKQ / 256, BATCH);
  softmax_col_kernel<<<gsm, blk, 0, stream>>>(attn);

  dim3 gcx(HIDDEN / 128, NKQ / 128, BATCH);
  context_gemm_kernel<<<gcx, blk, 0, stream>>>(attn, vb, ctx);
}
